// Model3_9070970929565
// MI455X (gfx1250) — matrix operand order repaired and checked by reading, not compiled, not run on hardware
//
#include <hip/hip_runtime.h>

// ---------------------------------------------------------------------------
// Fused 2-layer LSTM (H=50) + ReLU + FC head for MI455X (gfx1250, wave32).
// 128 persistent blocks (one 16-row batch tile each), 4 waves per block.
// Branch-free per-timestep path: gate tiles padded to 16 (4 per wave,
// unguarded), bias folded into the GEMM via a constant-1.0 A column, and
// pointwise overflow lanes write into zero-weight "dump" columns of A.
// Recurrent GEMMs: v_wmma_f32_16x16x32_f16, f32 accumulate, inline-0 C.
// ---------------------------------------------------------------------------

typedef __attribute__((ext_vector_type(16))) _Float16 v16h;
typedef __attribute__((ext_vector_type(8)))  _Float16 v8h;
typedef __attribute__((ext_vector_type(8)))  float    v8f;

#define SEQ   512
#define BATCH 2048
#define INARG 4
#define H     50
#define G4    200          // 4*H real gate columns
#define NTP   16           // PADDED gate tiles of 16 (256 cols; 13 real)
#define A1S   80           // layer-1 A row stride in halfs (K=64, 160B: 16B-aligned)
#define A2S   136          // layer-2 A row stride in halfs (K=128, 272B: 16B-aligned)

// A1 columns (K=64):  [ x: 0..3 | 1.0: 4 | 0: 5..7 | h1: 8..57 | dump: 58..63 ]
// A2 columns (K=128): [ h2: 0..49 | dump: 50..55 | relu(h1): 56..105
//                       | dump: 106..111 | 1.0: 112 | 0: 113..127 ]
// Wcat rows for dump/zero columns are 0, so garbage there contributes nothing.

struct __align__(32) SharedMem {
  // B fragments, exact WMMA B layout: frag = 32 lanes x 16 halfs per lane
  _Float16 w1[NTP * 2 * 32 * 16];  // layer1: 2 K-tiles  (32768 B)
  _Float16 w2[NTP * 4 * 32 * 16];  // layer2: 4 K-tiles  (65536 B)
  _Float16 A1[16 * A1S];           // 2560 B
  _Float16 A2[16 * A2S];           // 4352 B
  float    gates[NTP * 16 * 16];   // TRANSPOSED: [col 0..255][row 0..15] (16384 B)
  float    wfc[64];                // zero-padded beyond H
  float    outPart[16 * 8];
  float    bfc;
};

__device__ __forceinline__ float sigm(float v) {
  return 1.f / (1.f + __expf(-v));
}
__device__ __forceinline__ float tanh_fast(float v) {
  float e = __expf(fminf(-2.f * v, 80.f));   // clamp avoids inf/inf
  return (1.f - e) / (1.f + e);
}

__device__ __forceinline__ v8f wmma_f16(v16h a, v16h b, v8f c) {
  // (neg_a, A, neg_b, B, c_mod, C, reuse_a, reuse_b)
  return __builtin_amdgcn_wmma_f32_16x16x32_f16(false, a, false, b, (short)0, c,
                                                false, false);
}

// A-matrix fragment: lane<16 holds K {0..15} of row m=lane,
// lane>=16 holds K {16..31} of row m=lane-16. Two aligned 16B LDS reads.
__device__ __forceinline__ v16h load_a(const _Float16* base, int stride, int kt, int lane) {
  const int m   = lane & 15;
  const int off = (lane & 16) ? 16 : 0;
  const _Float16* p = base + m * stride + kt * 32 + off;
  v8h lo = *(const v8h*)(p);
  v8h hi = *(const v8h*)(p + 8);
  return __builtin_shufflevector(lo, hi, 0, 1, 2, 3, 4, 5, 6, 7,
                                 8, 9, 10, 11, 12, 13, 14, 15);
}

__global__ __launch_bounds__(128)
void lstm2_fused(const float* __restrict__ x,
                 const float* __restrict__ w_ih1, const float* __restrict__ w_hh1,
                 const float* __restrict__ b_ih1, const float* __restrict__ b_hh1,
                 const float* __restrict__ w_ih2, const float* __restrict__ w_hh2,
                 const float* __restrict__ b_ih2, const float* __restrict__ b_hh2,
                 const float* __restrict__ w_fc,  const float* __restrict__ b_fc,
                 float* __restrict__ out)
{
  extern __shared__ __align__(32) char smem_raw[];
  SharedMem& sm = *reinterpret_cast<SharedMem*>(smem_raw);

  const int tid    = threadIdx.x;      // 128 threads = 4 waves
  const int lane   = tid & 31;
  const int wave   = tid >> 5;
  const int row    = tid & 15;         // pointwise: batch row
  const int q      = tid >> 4;         // pointwise: hidden phase 0..7 (j = q + 8u)
  const int batch0 = blockIdx.x * 16;

  // ------------------------- one-time block init --------------------------
  for (int i = tid; i < 16 * A1S; i += 128) sm.A1[i] = (_Float16)0.f;
  for (int i = tid; i < 16 * A2S; i += 128) sm.A2[i] = (_Float16)0.f;
  if (tid < 64) sm.wfc[tid] = (tid < H) ? w_fc[tid] : 0.f;
  if (tid == 0) sm.bfc = b_fc[0];

  // Pack layer-1 B fragments:
  //   Wcat1[g][k] = w_ih1 (k<4) | bias1 (k==4) | w_hh1 (8<=k<58) | 0
  for (int idx = tid; idx < NTP * 2 * 32; idx += 128) {
    const int nt = idx >> 6;
    const int kt = (idx >> 5) & 1;
    const int l  = idx & 31;
    const int g  = nt * 16 + (l & 15);
    const int kbase = kt * 32 + ((l & 16) ? 16 : 0);
    _Float16* dst = &sm.w1[idx * 16];
    for (int v = 0; v < 16; ++v) {
      const int k = kbase + v;
      float val = 0.f;
      if (g < G4) {
        if (k < INARG)             val = w_ih1[g * INARG + k];
        else if (k == 4)           val = b_ih1[g] + b_hh1[g];
        else if (k >= 8 && k < 58) val = w_hh1[g * H + (k - 8)];
      }
      dst[v] = (_Float16)val;
    }
  }
  // Pack layer-2 B fragments:
  //   Wcat2[g][k] = w_hh2 (k<50) | w_ih2 (56<=k<106) | bias2 (k==112) | 0
  for (int idx = tid; idx < NTP * 4 * 32; idx += 128) {
    const int nt = idx >> 7;
    const int kt = (idx >> 5) & 3;
    const int l  = idx & 31;
    const int g  = nt * 16 + (l & 15);
    const int kbase = kt * 32 + ((l & 16) ? 16 : 0);
    _Float16* dst = &sm.w2[idx * 16];
    for (int v = 0; v < 16; ++v) {
      const int k = kbase + v;
      float val = 0.f;
      if (g < G4) {
        if (k < H)                   val = w_hh2[g * H + k];
        else if (k >= 56 && k < 106) val = w_ih2[g * H + (k - 56)];
        else if (k == 112)           val = b_ih2[g] + b_hh2[g];
      }
      dst[v] = (_Float16)val;
    }
  }

  // Constant-1.0 columns feed the bias rows of Wcat
  if (tid < 16) {
    sm.A1[tid * A1S + 4]   = (_Float16)1.f;
    sm.A2[tid * A2S + 112] = (_Float16)1.f;
  }

  // Cell state in registers: lane covers row `row`, hidden j = q + 8u
  float c1[7], c2[7];
#pragma unroll
  for (int u = 0; u < 7; ++u) { c1[u] = 0.f; c2[u] = 0.f; }

  __syncthreads();

  // ------------------------------ time loop -------------------------------
  for (int t = 0; t < SEQ; ++t) {
    // Phase A: stage x_t (f32 -> f16) into A1 cols 0..3; prefetch x_{t+1}
    // (past-the-end prefetch is speculative: silently dropped per ISA)
    if (tid < 16) {
      const float4 xv = *(const float4*)(x + ((size_t)t * BATCH + batch0 + tid) * INARG);
      _Float16* dst = &sm.A1[tid * A1S];
      dst[0] = (_Float16)xv.x; dst[1] = (_Float16)xv.y;
      dst[2] = (_Float16)xv.z; dst[3] = (_Float16)xv.w;
      __builtin_prefetch(x + ((size_t)(t + 1) * BATCH + batch0 + tid) * INARG, 0, 1);
    }
    __syncthreads();

    // Phase B: layer-1 GEMM, gates = A1 @ Wcat1^T (K=64, bias in-matrix)
    {
      const v16h a0 = load_a(sm.A1, A1S, 0, lane);
      const v16h a1 = load_a(sm.A1, A1S, 1, lane);
#pragma unroll
      for (int ntt = 0; ntt < 4; ++ntt) {
        const int nt = wave * 4 + ntt;                 // 16 padded tiles, no guard
        v8f acc = {};                                  // inline-0 C operand
        const v16h b0 = *(const v16h*)&sm.w1[(nt * 2 + 0) * 512 + lane * 16];
        const v16h b1 = *(const v16h*)&sm.w1[(nt * 2 + 1) * 512 + lane * 16];
        acc = wmma_f16(a0, b0, acc);
        acc = wmma_f16(a1, b1, acc);
        const int col = nt * 16 + (lane & 15);
        const int mb  = (lane & 16) ? 8 : 0;
        *(v8f*)&sm.gates[col * 16 + mb] = acc;         // 32B aligned, 2x ds_store_b128
      }
    }
    __syncthreads();

    // Phase C: layer-1 LSTM pointwise (branch-free; j>=50 lanes hit dump cols)
#pragma unroll
    for (int u = 0; u < 7; ++u) {
      const int j = q + 8 * u;                         // 0..55
      const float gi = sm.gates[(j)*16 + row];
      const float gf = sm.gates[(H + j) * 16 + row];
      const float gg = sm.gates[(2 * H + j) * 16 + row];
      const float go = sm.gates[(3 * H + j) * 16 + row];
      const float c = sigm(gf) * c1[u] + sigm(gi) * tanh_fast(gg);
      c1[u] = c;
      const float h = sigm(go) * tanh_fast(c);
      sm.A1[row * A1S + 8 + j]  = (_Float16)h;             // raw h for recurrence
      sm.A2[row * A2S + 56 + j] = (_Float16)fmaxf(h, 0.f); // relu(h1) feeds layer 2
    }
    __syncthreads();

    // Phase D: layer-2 GEMM, gates = A2 @ Wcat2^T (K=128, bias in-matrix)
    {
      const v16h a0 = load_a(sm.A2, A2S, 0, lane);
      const v16h a1 = load_a(sm.A2, A2S, 1, lane);
      const v16h a2 = load_a(sm.A2, A2S, 2, lane);
      const v16h a3 = load_a(sm.A2, A2S, 3, lane);
#pragma unroll
      for (int ntt = 0; ntt < 4; ++ntt) {
        const int nt = wave * 4 + ntt;
        v8f acc = {};
        const v16h b0 = *(const v16h*)&sm.w2[(nt * 4 + 0) * 512 + lane * 16];
        const v16h b1 = *(const v16h*)&sm.w2[(nt * 4 + 1) * 512 + lane * 16];
        const v16h b2 = *(const v16h*)&sm.w2[(nt * 4 + 2) * 512 + lane * 16];
        const v16h b3 = *(const v16h*)&sm.w2[(nt * 4 + 3) * 512 + lane * 16];
        acc = wmma_f16(a0, b0, acc);
        acc = wmma_f16(a1, b1, acc);
        acc = wmma_f16(a2, b2, acc);
        acc = wmma_f16(a3, b3, acc);
        const int col = nt * 16 + (lane & 15);
        const int mb  = (lane & 16) ? 8 : 0;
        *(v8f*)&sm.gates[col * 16 + mb] = acc;
      }
    }
    __syncthreads();

    // Phase E: layer-2 LSTM pointwise + FC partial (branch-free)
    {
      float dot = 0.f;
#pragma unroll
      for (int u = 0; u < 7; ++u) {
        const int j = q + 8 * u;                       // 0..55
        const float gi = sm.gates[(j)*16 + row];
        const float gf = sm.gates[(H + j) * 16 + row];
        const float gg = sm.gates[(2 * H + j) * 16 + row];
        const float go = sm.gates[(3 * H + j) * 16 + row];
        const float c = sigm(gf) * c2[u] + sigm(gi) * tanh_fast(gg);
        c2[u] = c;
        const float h = sigm(go) * tanh_fast(c);
        sm.A2[row * A2S + j] = (_Float16)h;            // h2 (j>=50 -> dump cols)
        dot += fmaxf(h, 0.f) * sm.wfc[j];              // wfc[j>=50] == 0
      }
      sm.outPart[row * 8 + q] = dot;
    }
    __syncthreads();

    // Phase F: reduce 8 partials per row, add fc bias, store out[t, batch, 0]
    if (tid < 16) {
      const float* p = &sm.outPart[tid * 8];
      const float r = ((p[0] + p[1]) + (p[2] + p[3])) +
                      ((p[4] + p[5]) + (p[6] + p[7])) + sm.bfc;
      out[(size_t)t * BATCH + batch0 + tid] = r;
    }
    __syncthreads();   // protect A1/A2/gates/outPart reuse next iteration
  }
}

extern "C" void kernel_launch(void* const* d_in, const int* in_sizes, int n_in,
                              void* d_out, int out_size, void* d_ws, size_t ws_size,
                              hipStream_t stream) {
  (void)in_sizes; (void)n_in; (void)out_size; (void)d_ws; (void)ws_size;
  const float* xp   = (const float*)d_in[0];
  const float* wih1 = (const float*)d_in[1];
  const float* whh1 = (const float*)d_in[2];
  const float* bih1 = (const float*)d_in[3];
  const float* bhh1 = (const float*)d_in[4];
  const float* wih2 = (const float*)d_in[5];
  const float* whh2 = (const float*)d_in[6];
  const float* bih2 = (const float*)d_in[7];
  const float* bhh2 = (const float*)d_in[8];
  const float* wfc  = (const float*)d_in[9];
  const float* bfc  = (const float*)d_in[10];
  float* out = (float*)d_out;

  dim3 grid(BATCH / 16);   // 128 persistent blocks, one 16-row batch tile each
  dim3 block(128);         // 4 waves, 4 gate tiles each (balanced, unguarded)
  size_t shmem = sizeof(SharedMem);   // ~122 KB dynamic LDS (WGP has 320 KB)
  lstm2_fused<<<grid, block, shmem, stream>>>(xp, wih1, whh1, bih1, bhh1,
                                              wih2, whh2, bih2, bhh2,
                                              wfc, bfc, out);
}
